// CausalMultiHeadAttention_82832739271252
// MI455X (gfx1250) — matrix operand order repaired and checked by reading, not compiled, not run on hardware
//
#include <hip/hip_runtime.h>
#include <hip/hip_bf16.h>

// ---------------- CDNA5 (gfx1250) causal MHA, bf16 WMMA path ----------------
typedef __bf16 bf16_t;
typedef __attribute__((ext_vector_type(16))) __bf16 v16bf;
typedef __attribute__((ext_vector_type(8)))  float  v8f;

#define Bn 4
#define Sn 2048
#define Cn 768
#define Hn 6
#define Dn 128
#define Mn (Bn * Sn)      // 8192 rows
#define NQKV (3 * Cn)     // 2304

__device__ __forceinline__ v8f zero8() {
  v8f z;
#pragma unroll
  for (int i = 0; i < 8; ++i) z[i] = 0.0f;
  return z;
}

__device__ __forceinline__ v8f wmma_bf16(v16bf a, v16bf b, v8f c) {
  // v_wmma_f32_16x16x32_bf16 : D = A(16x32) * B(32x16) + C(16x16 f32)
  return __builtin_amdgcn_wmma_f32_16x16x32_bf16(false, a, false, b,
                                                 (short)0, c, false, false);
}

// A fragment (16x32 bf16, row-major source, leading dim ld).
__device__ __forceinline__ v16bf load_a16(const bf16_t* p, int ld) {
  const int lane = threadIdx.x & 31;
  const int row  = lane & 15;
  const int kb   = (lane & 16) ? 16 : 0;
  const bf16_t* r0 = p + row * ld + kb;
  v16bf a;
#pragma unroll
  for (int j = 0; j < 8; ++j) a[j] = r0[j];
#pragma unroll
  for (int j = 0; j < 8; ++j) a[8 + j] = r0[8 + j];
  return a;
}

// B fragment (32x16 bf16) from a COLUMN-MAJOR source: element (k,n) at p[n*ld + k].
__device__ __forceinline__ v16bf load_bT(const bf16_t* p, int ld) {
  const int lane = threadIdx.x & 31;
  const int col  = lane & 15;
  const int kb   = (lane & 16) ? 16 : 0;
  const bf16_t* r0 = p + col * ld + kb;
  v16bf b;
#pragma unroll
  for (int j = 0; j < 16; ++j) b[j] = r0[j];
  return b;
}

// -------------------------- conversion kernels ------------------------------
__global__ void cvt_f32_bf16(const float* __restrict__ src,
                             bf16_t* __restrict__ dst, int n) {
  int i = blockIdx.x * blockDim.x + threadIdx.x;
  if (i < n) dst[i] = (bf16_t)src[i];
}

// src is [K x N] row-major fp32 ; dst is [N x K] row-major bf16 (transposed)
__global__ void cvt_transpose_bf16(const float* __restrict__ src,
                                   bf16_t* __restrict__ dst, int K, int N) {
  int i = blockIdx.x * blockDim.x + threadIdx.x;
  if (i < K * N) {
    int k = i % K, n = i / K;
    dst[i] = (bf16_t)src[k * N + n];
  }
}

// ------------------------------ QKV GEMM ------------------------------------
// One wave per 16x64 tile, software-pipelined (double-buffered fragments).
__global__ __launch_bounds__(256) void qkv_gemm_kernel(
    const bf16_t* __restrict__ xb,    // [M, C]
    const bf16_t* __restrict__ wT,    // [3C, C]
    const float*  __restrict__ bias,  // [3C]
    bf16_t* __restrict__ qb, bf16_t* __restrict__ kbuf,
    bf16_t* __restrict__ vTb) {
  const int wave = __builtin_amdgcn_readfirstlane(
      blockIdx.x * (blockDim.x >> 5) + (threadIdx.x >> 5));
  const int Nt = NQKV / 64;                 // 36
  const int total = (Mn / 16) * Nt;
  if (wave >= total) return;                // scalar branch
  const int m0 = (wave / Nt) * 16;
  const int n0 = (wave % Nt) * 64;

  v8f acc[4];
#pragma unroll
  for (int t = 0; t < 4; ++t) acc[t] = zero8();

  const bf16_t* ap = xb + m0 * Cn;
  const bf16_t* bp = wT + n0 * Cn;

  // prologue loads
  v16bf a  = load_a16(ap, Cn);
  v16bf b0 = load_bT(bp + 0 * 16 * Cn, Cn);
  v16bf b1 = load_bT(bp + 1 * 16 * Cn, Cn);
  v16bf b2 = load_bT(bp + 2 * 16 * Cn, Cn);
  v16bf b3 = load_bT(bp + 3 * 16 * Cn, Cn);

  for (int k0 = 0; k0 < Cn - 32; k0 += 32) {
    // issue next-iteration loads BEFORE this iteration's WMMAs
    v16bf an  = load_a16(ap + k0 + 32, Cn);
    v16bf bn0 = load_bT(bp + 0 * 16 * Cn + k0 + 32, Cn);
    v16bf bn1 = load_bT(bp + 1 * 16 * Cn + k0 + 32, Cn);
    v16bf bn2 = load_bT(bp + 2 * 16 * Cn + k0 + 32, Cn);
    v16bf bn3 = load_bT(bp + 3 * 16 * Cn + k0 + 32, Cn);
    acc[0] = wmma_bf16(a, b0, acc[0]);
    acc[1] = wmma_bf16(a, b1, acc[1]);
    acc[2] = wmma_bf16(a, b2, acc[2]);
    acc[3] = wmma_bf16(a, b3, acc[3]);
    a = an; b0 = bn0; b1 = bn1; b2 = bn2; b3 = bn3;
  }
  acc[0] = wmma_bf16(a, b0, acc[0]);
  acc[1] = wmma_bf16(a, b1, acc[1]);
  acc[2] = wmma_bf16(a, b2, acc[2]);
  acc[3] = wmma_bf16(a, b3, acc[3]);

  // epilogue: which (q/k/v) and head are uniform per wave (64 | 128 | 768)
  const int lane   = threadIdx.x & 31;
  const int lane15 = lane & 15;
  const int which  = n0 / Cn;
  const int cbase  = n0 % Cn;
  const int h      = cbase / Dn;
  const int dbase  = cbase % Dn;
  const int bb = m0 / Sn;
  const int sb = (m0 % Sn) + ((lane & 16) ? 8 : 0);
#pragma unroll
  for (int t = 0; t < 4; ++t) {
    const int d  = dbase + t * 16 + lane15;
    const float bv = bias[n0 + t * 16 + lane15];
#pragma unroll
    for (int r = 0; r < 8; ++r) {
      const int s = sb + r;
      const bf16_t val = (bf16_t)(acc[t][r] + bv);
      if (which == 0)      qb  [(((bb * Hn) + h) * Sn + s) * Dn + d] = val;
      else if (which == 1) kbuf[(((bb * Hn) + h) * Sn + s) * Dn + d] = val;
      else                 vTb [(((bb * Hn) + h) * Dn + d) * Sn + s] = val;
    }
  }
}

// --------------------------- Flash attention --------------------------------
__global__ __launch_bounds__(256) void attn_kernel(
    const bf16_t* __restrict__ qb,    // [B,H,S,D]
    const bf16_t* __restrict__ kbuf,  // [B,H,S,D]
    const bf16_t* __restrict__ vTb,   // [B,H,D,S]
    bf16_t* __restrict__ ob) {        // [B,S,C] bf16
  __shared__ bf16_t sP[8][16 * 32];   // per-wave P staging (1KB each)
  const int wslot = threadIdx.x >> 5;
  const int wave  = __builtin_amdgcn_readfirstlane(blockIdx.x * 8 + wslot);
  const int QT = Sn / 16;             // 128
  const int bh = wave / QT;
  const int qt = wave % QT;
  const int s0 = qt * 16;
  const int lane = threadIdx.x & 31;
  const int half = (lane & 16) ? 8 : 0;

  const bf16_t* Q  = qb   + (size_t)bh * Sn * Dn;
  const bf16_t* Kp = kbuf + (size_t)bh * Sn * Dn;
  const bf16_t* VT = vTb  + (size_t)bh * Dn * Sn;

  v16bf qf[4];
#pragma unroll
  for (int kc = 0; kc < 4; ++kc) qf[kc] = load_a16(Q + s0 * Dn + kc * 32, Dn);

  v8f acc[8];
#pragma unroll
  for (int n = 0; n < 8; ++n) acc[n] = zero8();
  float mrow[8], lrow[8];
#pragma unroll
  for (int r = 0; r < 8; ++r) { mrow[r] = -1e30f; lrow[r] = 0.0f; }

  const float sm_scale = 0.08838834764831845f;  // 1/sqrt(128)
  bf16_t* sp = sP[wslot];
  const int s_hi = s0 + 15;

  for (int t0 = 0; t0 <= s_hi; t0 += 32) {      // wave-uniform trip count
    // ---- scores: Q(16x128) . K^T(128x32), K-fragments pipelined over kc
    v8f sa0 = zero8(), sa1 = zero8();
    v16bf kb0 = load_bT(Kp + t0 * Dn, Dn);
    v16bf kb1 = load_bT(Kp + (t0 + 16) * Dn, Dn);
#pragma unroll
    for (int kc = 0; kc < 4; ++kc) {
      v16bf kn0, kn1;
      if (kc < 3) {
        kn0 = load_bT(Kp + t0 * Dn + (kc + 1) * 32, Dn);
        kn1 = load_bT(Kp + (t0 + 16) * Dn + (kc + 1) * 32, Dn);
      }
      sa0 = wmma_bf16(qf[kc], kb0, sa0);
      sa1 = wmma_bf16(qf[kc], kb1, sa1);
      if (kc < 3) { kb0 = kn0; kb1 = kn1; }
    }

    // ---- scale + causal mask (row = s0+half+r, col = lane%16)
    float p0[8], p1[8];
    const int col0 = t0 + (lane & 15);
    const int col1 = col0 + 16;
#pragma unroll
    for (int r = 0; r < 8; ++r) {
      const int row = s0 + half + r;
      float x0 = sa0[r] * sm_scale;
      float x1 = sa1[r] * sm_scale;
      p0[r] = (col0 > row) ? -1e30f : x0;
      p1[r] = (col1 > row) ? -1e30f : x1;
    }

    // ---- online softmax: rows live in 16-lane half-waves
#pragma unroll
    for (int r = 0; r < 8; ++r) {
      float mx = fmaxf(p0[r], p1[r]);
#pragma unroll
      for (int msk = 8; msk >= 1; msk >>= 1)
        mx = fmaxf(mx, __shfl_xor(mx, msk, 16));
      const float mnew = fmaxf(mrow[r], mx);
      const float corr = __expf(mrow[r] - mnew);
      mrow[r] = mnew;
      const float e0 = __expf(p0[r] - mnew);
      const float e1 = __expf(p1[r] - mnew);
      p0[r] = e0; p1[r] = e1;
      float rs = e0 + e1;
#pragma unroll
      for (int msk = 8; msk >= 1; msk >>= 1)
        rs += __shfl_xor(rs, msk, 16);
      lrow[r] = lrow[r] * corr + rs;
#pragma unroll
      for (int n = 0; n < 8; ++n) acc[n][r] *= corr;
    }

    // ---- restage P (accumulator layout -> A-fragment layout) via private LDS
#pragma unroll
    for (int r = 0; r < 8; ++r) {
      sp[(half + r) * 32 + (lane & 15)]      = (bf16_t)p0[r];
      sp[(half + r) * 32 + 16 + (lane & 15)] = (bf16_t)p1[r];
    }
    v16bf pa = load_a16(sp, 32);   // same-wave DS ops are in-order

    // ---- O += P(16x32) . V(32x128), V-fragments pipelined in pairs
    v16bf vb0 = load_bT(VT + 0 * 16 * Sn + t0, Sn);
    v16bf vb1 = load_bT(VT + 1 * 16 * Sn + t0, Sn);
#pragma unroll
    for (int n = 0; n < 8; n += 2) {
      v16bf vn0, vn1;
      if (n < 6) {
        vn0 = load_bT(VT + (n + 2) * 16 * Sn + t0, Sn);
        vn1 = load_bT(VT + (n + 3) * 16 * Sn + t0, Sn);
      }
      acc[n]     = wmma_bf16(pa, vb0, acc[n]);
      acc[n + 1] = wmma_bf16(pa, vb1, acc[n + 1]);
      if (n < 6) { vb0 = vn0; vb1 = vn1; }
    }
  }

  // ---- normalize + store to [B,S,C] bf16
  const int bb = bh / Hn, h = bh % Hn;
#pragma unroll
  for (int r = 0; r < 8; ++r) {
    const float inv = 1.0f / lrow[r];
    const int srow = s0 + half + r;
#pragma unroll
    for (int n = 0; n < 8; ++n) {
      ob[(size_t)(bb * Sn + srow) * Cn + h * Dn + n * 16 + (lane & 15)] =
          (bf16_t)(acc[n][r] * inv);
    }
  }
}

// ------------------------------ proj GEMM -----------------------------------
__global__ __launch_bounds__(256) void proj_gemm_kernel(
    const bf16_t* __restrict__ aob,   // [M, C] bf16
    const bf16_t* __restrict__ wT,    // [C, C]
    const float*  __restrict__ bias,  // [C]
    float* __restrict__ out) {        // [M, C] fp32
  const int wave = __builtin_amdgcn_readfirstlane(
      blockIdx.x * (blockDim.x >> 5) + (threadIdx.x >> 5));
  const int Nt = Cn / 64;             // 12
  const int total = (Mn / 16) * Nt;
  if (wave >= total) return;
  const int m0 = (wave / Nt) * 16;
  const int n0 = (wave % Nt) * 64;

  v8f acc[4];
#pragma unroll
  for (int t = 0; t < 4; ++t) acc[t] = zero8();

  const bf16_t* ap = aob + m0 * Cn;
  const bf16_t* bp = wT + n0 * Cn;

  v16bf a  = load_a16(ap, Cn);
  v16bf b0 = load_bT(bp + 0 * 16 * Cn, Cn);
  v16bf b1 = load_bT(bp + 1 * 16 * Cn, Cn);
  v16bf b2 = load_bT(bp + 2 * 16 * Cn, Cn);
  v16bf b3 = load_bT(bp + 3 * 16 * Cn, Cn);

  for (int k0 = 0; k0 < Cn - 32; k0 += 32) {
    v16bf an  = load_a16(ap + k0 + 32, Cn);
    v16bf bn0 = load_bT(bp + 0 * 16 * Cn + k0 + 32, Cn);
    v16bf bn1 = load_bT(bp + 1 * 16 * Cn + k0 + 32, Cn);
    v16bf bn2 = load_bT(bp + 2 * 16 * Cn + k0 + 32, Cn);
    v16bf bn3 = load_bT(bp + 3 * 16 * Cn + k0 + 32, Cn);
    acc[0] = wmma_bf16(a, b0, acc[0]);
    acc[1] = wmma_bf16(a, b1, acc[1]);
    acc[2] = wmma_bf16(a, b2, acc[2]);
    acc[3] = wmma_bf16(a, b3, acc[3]);
    a = an; b0 = bn0; b1 = bn1; b2 = bn2; b3 = bn3;
  }
  acc[0] = wmma_bf16(a, b0, acc[0]);
  acc[1] = wmma_bf16(a, b1, acc[1]);
  acc[2] = wmma_bf16(a, b2, acc[2]);
  acc[3] = wmma_bf16(a, b3, acc[3]);

  const int lane = threadIdx.x & 31;
  const int rbase = m0 + ((lane & 16) ? 8 : 0);
#pragma unroll
  for (int t = 0; t < 4; ++t) {
    const int col = n0 + t * 16 + (lane & 15);
    const float bv = bias[col];
#pragma unroll
    for (int r = 0; r < 8; ++r)
      out[(size_t)(rbase + r) * Cn + col] = acc[t][r] + bv;
  }
}

// ------------------------------ launcher ------------------------------------
extern "C" void kernel_launch(void* const* d_in, const int* in_sizes, int n_in,
                              void* d_out, int out_size, void* d_ws, size_t ws_size,
                              hipStream_t stream) {
  const float* x      = (const float*)d_in[0];  // [B,S,C]
  const float* w_attn = (const float*)d_in[1];  // [C,3C]
  const float* b_attn = (const float*)d_in[2];  // [3C]
  const float* w_proj = (const float*)d_in[3];  // [C,C]
  const float* b_proj = (const float*)d_in[4];  // [C]
  float* out = (float*)d_out;

  char* ws = (char*)d_ws;
  size_t off = 0;
  auto alloc = [&](size_t bytes) {
    char* p = ws + off;
    off += (bytes + 255) & ~(size_t)255;
    return p;
  };
  bf16_t* xb    = (bf16_t*)alloc((size_t)Mn * Cn * 2);
  bf16_t* wqT   = (bf16_t*)alloc((size_t)NQKV * Cn * 2);
  bf16_t* wpT   = (bf16_t*)alloc((size_t)Cn * Cn * 2);
  bf16_t* qbuf  = (bf16_t*)alloc((size_t)Mn * Cn * 2);
  bf16_t* kbuf  = (bf16_t*)alloc((size_t)Mn * Cn * 2);
  bf16_t* vTbuf = (bf16_t*)alloc((size_t)Mn * Cn * 2);
  bf16_t* aout  = (bf16_t*)alloc((size_t)Mn * Cn * 2);
  (void)ws_size;

  // 1. precision conversion (+ weight transposes for contiguous B-fragments)
  {
    int n = Mn * Cn;
    cvt_f32_bf16<<<(n + 255) / 256, 256, 0, stream>>>(x, xb, n);
    n = Cn * NQKV;
    cvt_transpose_bf16<<<(n + 255) / 256, 256, 0, stream>>>(w_attn, wqT, Cn, NQKV);
    n = Cn * Cn;
    cvt_transpose_bf16<<<(n + 255) / 256, 256, 0, stream>>>(w_proj, wpT, Cn, Cn);
  }

  // 2. fused QKV projection (bias + head split + V transpose)
  {
    int waves = (Mn / 16) * (NQKV / 64);     // 18432
    qkv_gemm_kernel<<<waves / 8, 256, 0, stream>>>(xb, wqT, b_attn, qbuf, kbuf, vTbuf);
  }

  // 3. flash attention (causal, online softmax)
  {
    int waves = Bn * Hn * (Sn / 16);         // 3072
    attn_kernel<<<waves / 8, 256, 0, stream>>>(qbuf, kbuf, vTbuf, aout);
  }

  // 4. output projection (bias fused, fp32 out)
  {
    int waves = (Mn / 16) * (Cn / 64);       // 6144
    proj_gemm_kernel<<<waves / 8, 256, 0, stream>>>(aout, wpT, b_proj, out);
  }
}
